// TitanMAG_42923903156556
// MI455X (gfx1250) — matrix operand order repaired and checked by reading, not compiled, not run on hardware
//
#include <hip/hip_runtime.h>

// ---------------------------------------------------------------------------
// CDNA5 WMMA (wave32): D(f32 16x16) = A(bf16 16x32) * B(bf16 32x16) + C
// ---------------------------------------------------------------------------
typedef __bf16 bf16;
typedef __bf16 v16bf __attribute__((ext_vector_type(16)));
typedef __bf16 v8bf  __attribute__((ext_vector_type(8)));
typedef float  v8f   __attribute__((ext_vector_type(8)));

constexpr int CDIM = 1024;   // model dim
constexpr int CHID = 4096;   // MLP hidden
constexpr int CNH  = 4;      // heads
constexpr int CDH  = 256;    // head dim
constexpr int CWIN = 512;    // attention window
constexpr int CPM  = 64;     // persistent memory tokens
constexpr int CB   = 2;      // batch
constexpr int CS   = 2048;   // seq
constexpr int CT   = CPM + CS;   // 2112 rows per batch
constexpr int CN   = CB * CT;    // 4224 total rows

static __device__ __forceinline__ v8f wmma_bf16(v16bf a, v16bf b, v8f c) {
  return __builtin_amdgcn_wmma_f32_16x16x32_bf16(false, a, false, b, (short)0, c,
                                                 false, false);
}
static __device__ __forceinline__ v16bf cat8(v8bf lo, v8bf hi) {
  return __builtin_shufflevector(lo, hi, 0, 1, 2, 3, 4, 5, 6, 7, 8, 9, 10, 11,
                                 12, 13, 14, 15);
}

// ---------------------------------------------------------------------------
// f32 -> bf16 convert
// ---------------------------------------------------------------------------
__global__ void cvt_f32_bf16(const float* __restrict__ src, bf16* __restrict__ dst,
                             int n) {
  int i = blockIdx.x * blockDim.x + threadIdx.x;
  if (i < n) dst[i] = (bf16)src[i];
}

// ---------------------------------------------------------------------------
// Build x_in = concat(persistent_mem, x) per batch, in bf16.  [CN, CDIM]
// ---------------------------------------------------------------------------
__global__ void prep_xin_kernel(const float* __restrict__ x,
                                const float* __restrict__ pmem,
                                bf16* __restrict__ xin) {
  int i = blockIdx.x * blockDim.x + threadIdx.x;
  if (i >= CN * CDIM) return;
  int n = i / CDIM, d = i - n * CDIM;
  int b = n / CT, t = n - b * CT;
  float v = (t < CPM) ? pmem[t * CDIM + d]
                      : x[((size_t)b * CS + (t - CPM)) * CDIM + d];
  xin[i] = (bf16)v;
}

// ---------------------------------------------------------------------------
// V transpose: [CN, CDIM] (col = h*CDH+d)  ->  vT[(b*CNH+h)*CDH + d][CT]
// so PV B-fragments read 16 consecutive keys contiguously.
// ---------------------------------------------------------------------------
__global__ void vtrans_kernel(const bf16* __restrict__ v, bf16* __restrict__ vt) {
  int i = blockIdx.x * blockDim.x + threadIdx.x;
  if (i >= CN * CDIM) return;
  int t = i % CT;
  int rem = i / CT;
  int d = rem % CDH;
  int bh = rem / CDH;            // b*CNH + h
  int b = bh / CNH, h = bh % CNH;
  vt[i] = v[((size_t)b * CT + t) * CDIM + h * CDH + d];
}

// ---------------------------------------------------------------------------
// Tiled bf16 WMMA GEMM: C[M,N] = op(A)*op(B), block tile 64x256, K-step 32.
// 8 waves in a 2x4 grid; each wave owns 32x64 (2x4 WMMA accumulators).
// LDS: A row-major [64][32]; B stored K-major per column [256][32] so every
// fragment load is 128-bit. Global->LDS staging is register double-buffered;
// all global staging reads are 16B-contiguous in both transpose modes.
//
// EPI: 0 f32 out (+opt bias)        1 bf16 out (+opt bias)
//      2 f32+bf16 out (+opt bias)   3 h=acc+bias -> Cb2=h, Cb=silu(h)
//      4 Cb = acc + bias - auxf     5 Cb = acc * silu'(auxb)
//      6 Cb = (1-alpha)*auxf - theta*acc   (fast-weight update)
//      7 Cb = silu(acc + bias)
// ---------------------------------------------------------------------------
template <int TA, int TB, int EPI>
__global__ __launch_bounds__(256) void gemm_bf16_kernel(
    const bf16* __restrict__ A, const bf16* __restrict__ B,
    float* __restrict__ Cf, bf16* __restrict__ Cb, bf16* __restrict__ Cb2,
    const float* __restrict__ bias, const float* __restrict__ auxf,
    const bf16* __restrict__ auxb, const float* __restrict__ alpha_p,
    const float* __restrict__ theta_p, int M, int N, int K, int lda, int ldb) {
  __shared__ alignas(32) bf16 As[64 * 32];
  __shared__ alignas(32) bf16 Bs[256 * 32];
  const int tid = threadIdx.x;
  const int lane = tid & 31, wave = tid >> 5;
  const int l16 = lane & 15, hf = lane >> 4;
  const int wr = wave >> 2, wc = wave & 3;  // 2x4 wave grid
  const int m0 = blockIdx.y * 64, n0 = blockIdx.x * 256;
  const int mw = wr * 32, nw = wc * 64;

  v8f c[2][4];
#pragma unroll
  for (int i = 0; i < 2; ++i)
#pragma unroll
    for (int j = 0; j < 4; ++j)
#pragma unroll
      for (int r = 0; r < 8; ++r) c[i][j][r] = 0.f;

  v8bf ra;      // 8 bf16 of the A tile per thread
  v8bf rb[4];   // 32 bf16 of the B tile per thread

  auto loadA = [&](int k0) {
    if constexpr (TA == 0) {
      int m = tid >> 2, kc = (tid & 3) * 8;
      ra = *(const v8bf*)&A[(size_t)(m0 + m) * lda + k0 + kc];
    } else {
      int kk = tid >> 3, mc = (tid & 7) * 8;
      ra = *(const v8bf*)&A[(size_t)(k0 + kk) * lda + m0 + mc];
    }
  };
  auto storeA = [&]() {
    if constexpr (TA == 0) {
      int m = tid >> 2, kc = (tid & 3) * 8;
      *(v8bf*)&As[m * 32 + kc] = ra;
    } else {
      int kk = tid >> 3, mc = (tid & 7) * 8;
#pragma unroll
      for (int j = 0; j < 8; ++j) As[(mc + j) * 32 + kk] = ra[j];
    }
  };
  auto loadB = [&](int k0) {
#pragma unroll
    for (int cc = 0; cc < 4; ++cc) {
      int ch = tid + 256 * cc;  // 0..1023
      if constexpr (TB == 1) {
        int n = ch >> 2, kc = (ch & 3) * 8;
        rb[cc] = *(const v8bf*)&B[(size_t)(n0 + n) * ldb + k0 + kc];
      } else {
        int kk = ch >> 5, nc = (ch & 31) * 8;
        rb[cc] = *(const v8bf*)&B[(size_t)(k0 + kk) * ldb + n0 + nc];
      }
    }
  };
  auto storeB = [&]() {
#pragma unroll
    for (int cc = 0; cc < 4; ++cc) {
      int ch = tid + 256 * cc;
      if constexpr (TB == 1) {
        int n = ch >> 2, kc = (ch & 3) * 8;
        *(v8bf*)&Bs[n * 32 + kc] = rb[cc];
      } else {
        int kk = ch >> 5, nc = (ch & 31) * 8;
#pragma unroll
        for (int j = 0; j < 8; ++j) Bs[(nc + j) * 32 + kk] = rb[cc][j];
      }
    }
  };

  loadA(0);
  loadB(0);
  storeA();
  storeB();
  __syncthreads();

  for (int k0 = 0; k0 < K; k0 += 32) {
    const bool has_next = (k0 + 32 < K);
    if (has_next) {  // register double-buffer: fetch next tile while computing
      loadA(k0 + 32);
      loadB(k0 + 32);
    }
    if (k0 + 64 < K) {  // global_prefetch_b8 for the tile after next
      __builtin_prefetch(TA ? &A[(size_t)(k0 + 64 + (tid >> 3)) * lda + m0]
                            : &A[(size_t)(m0 + (tid >> 2)) * lda + k0 + 64],
                         0, 1);
      __builtin_prefetch(TB ? &B[(size_t)(n0 + (tid >> 2)) * ldb + k0 + 64]
                            : &B[(size_t)(k0 + 64 + (tid >> 5)) * ldb + n0],
                         0, 1);
    }
    // ---- compute from LDS: all fragment loads are 128-bit ----
    v16bf av[2];
#pragma unroll
    for (int s = 0; s < 2; ++s) {
      int row = mw + s * 16 + l16;
      av[s] = *(const v16bf*)&As[row * 32 + hf * 16];
    }
#pragma unroll
    for (int j = 0; j < 4; ++j) {
      int col = nw + j * 16 + l16;
      v16bf bv = *(const v16bf*)&Bs[col * 32 + hf * 16];
#pragma unroll
      for (int i = 0; i < 2; ++i) c[i][j] = wmma_bf16(av[i], bv, c[i][j]);
    }
    __syncthreads();
    if (has_next) {
      storeA();
      storeB();
    }
    __syncthreads();
  }

  float alpha = 0.f, theta = 0.f;
  if (EPI == 6) { alpha = alpha_p[0]; theta = theta_p[0]; }

#pragma unroll
  for (int i = 0; i < 2; ++i)
#pragma unroll
    for (int j = 0; j < 4; ++j)
#pragma unroll
      for (int r = 0; r < 8; ++r) {
        int row = m0 + mw + i * 16 + r + 8 * hf;  // C layout: VGPR r, half hf
        int col = n0 + nw + j * 16 + l16;
        size_t idx = (size_t)row * N + col;
        float acc = c[i][j][r];
        if (EPI == 0) {
          if (bias) acc += bias[col];
          Cf[idx] = acc;
        } else if (EPI == 1) {
          if (bias) acc += bias[col];
          Cb[idx] = (bf16)acc;
        } else if (EPI == 2) {
          if (bias) acc += bias[col];
          Cf[idx] = acc;
          Cb[idx] = (bf16)acc;
        } else if (EPI == 3) {
          float hh = acc + bias[col];
          Cb2[idx] = (bf16)hh;
          float s = 1.f / (1.f + __expf(-hh));
          Cb[idx] = (bf16)(hh * s);
        } else if (EPI == 4) {
          float ev = acc + bias[col] - auxf[idx];
          Cb[idx] = (bf16)ev;
        } else if (EPI == 5) {
          float hh = (float)auxb[idx];
          float s = 1.f / (1.f + __expf(-hh));
          float ds = s * (1.f + hh * (1.f - s));
          Cb[idx] = (bf16)(acc * ds);
        } else if (EPI == 6) {
          Cb[idx] = (bf16)((1.f - alpha) * auxf[idx] - theta * acc);
        } else if (EPI == 7) {
          float hh = acc + bias[col];
          float s = 1.f / (1.f + __expf(-hh));
          Cb[idx] = (bf16)(hh * s);
        }
      }
}

// ---------------------------------------------------------------------------
// Sliding-window flash attention, one wave per 16-query tile, 32-key tiles.
// q,k stored [CN, CDIM] (col = h*CDH+d); v pre-transposed to [bh, dh, CT].
// ---------------------------------------------------------------------------
__global__ __launch_bounds__(128) void flash_attn_kernel(
    const bf16* __restrict__ qg, const bf16* __restrict__ kg,
    const bf16* __restrict__ vt, bf16* __restrict__ og) {
  __shared__ alignas(32) bf16 p_s[4][16 * 32];  // per-wave P staging
  const int lane = threadIdx.x & 31, wave = threadIdx.x >> 5;
  const int l16 = lane & 15, hf = lane >> 4;
  const int q0 = blockIdx.x * 64 + wave * 16;
  const int b = blockIdx.y >> 2, h = blockIdx.y & 3;
  const size_t rowbase = (size_t)b * CT;
  const int hoff = h * CDH;

  // Q fragments: 16 rows x 256 dims -> 8 A-fragments (one contiguous 32B load each)
  v16bf qf[8];
  {
    const bf16* qptr = qg + (rowbase + q0 + l16) * CDIM + hoff;
#pragma unroll
    for (int kk = 0; kk < 8; ++kk) {
      qf[kk] = *(const v16bf*)(qptr + kk * 32 + hf * 16);
    }
  }

  v8f o[16];
#pragma unroll
  for (int j = 0; j < 16; ++j)
#pragma unroll
    for (int r = 0; r < 8; ++r) o[j][r] = 0.f;
  float m_i[8], l_i[8];
#pragma unroll
  for (int r = 0; r < 8; ++r) { m_i[r] = -1e30f; l_i[r] = 0.f; }

  int kt0 = q0 - (CWIN - 1);
  if (kt0 < 0) kt0 = 0;
  kt0 &= ~31;
  bf16* ps = p_s[wave];
  const bf16* vtb = vt + (size_t)(b * CNH + h) * CDH * CT;

  for (int kt = kt0; kt <= q0 + 15; kt += 32) {
    // ---- S = Q * K^T (16 x 32): K fragments are contiguous 32B loads ----
    v8f sfr[2];
#pragma unroll
    for (int n0 = 0; n0 < 2; ++n0) {
#pragma unroll
      for (int r = 0; r < 8; ++r) sfr[n0][r] = 0.f;
      const bf16* kptr =
          kg + (rowbase + kt + n0 * 16 + l16) * CDIM + hoff + hf * 16;
#pragma unroll
      for (int kk = 0; kk < 8; ++kk) {
        v16bf bfr = *(const v16bf*)(kptr + kk * 32);
        sfr[n0] = wmma_bf16(qf[kk], bfr, sfr[n0]);
      }
    }
    // ---- banded causal mask + scale (1/sqrt(256)) ----
#pragma unroll
    for (int n0 = 0; n0 < 2; ++n0)
#pragma unroll
      for (int r = 0; r < 8; ++r) {
        int irow = q0 + r + 8 * hf;
        int jcol = kt + n0 * 16 + l16;
        bool ok = (jcol <= irow) && (jcol > irow - CWIN);
        sfr[n0][r] = ok ? sfr[n0][r] * 0.0625f : -1e30f;
      }
    // ---- online softmax (row reductions across 16-lane halves) ----
#pragma unroll
    for (int r = 0; r < 8; ++r) {
      float mx = fmaxf(sfr[0][r], sfr[1][r]);
#pragma unroll
      for (int msk = 8; msk >= 1; msk >>= 1) mx = fmaxf(mx, __shfl_xor(mx, msk));
      float mnew = fmaxf(m_i[r], mx);
      bool any = mnew > -1e29f;
      float p0 = any ? __expf(sfr[0][r] - mnew) : 0.f;
      float p1 = any ? __expf(sfr[1][r] - mnew) : 0.f;
      float rs = p0 + p1;
#pragma unroll
      for (int msk = 8; msk >= 1; msk >>= 1) rs += __shfl_xor(rs, msk);
      float corr = any ? __expf(m_i[r] - mnew) : 1.f;
      l_i[r] = l_i[r] * corr + rs;
      m_i[r] = mnew;
#pragma unroll
      for (int j = 0; j < 16; ++j) o[j][r] *= corr;
      sfr[0][r] = p0;
      sfr[1][r] = p1;
    }
    // ---- P: C-layout -> LDS -> A-fragment layout ----
#pragma unroll
    for (int n0 = 0; n0 < 2; ++n0)
#pragma unroll
      for (int r = 0; r < 8; ++r)
        ps[(r + 8 * hf) * 32 + n0 * 16 + l16] = (bf16)sfr[n0][r];
    asm volatile("s_wait_dscnt 0" ::: "memory");
    v16bf pf = *(const v16bf*)&ps[l16 * 32 + hf * 16];
    asm volatile("s_wait_dscnt 0" ::: "memory");
    // ---- O += P * V: V fragments contiguous via transposed V ----
#pragma unroll
    for (int j = 0; j < 16; ++j) {
      v16bf vfr =
          *(const v16bf*)(vtb + (size_t)(j * 16 + l16) * CT + kt + hf * 16);
      o[j] = wmma_bf16(pf, vfr, o[j]);
    }
  }

  // ---- normalize and write per-head output back to [CN, CDIM] ----
#pragma unroll
  for (int j = 0; j < 16; ++j)
#pragma unroll
    for (int r = 0; r < 8; ++r) {
      float val = o[j][r] / l_i[r];
      og[(rowbase + q0 + r + 8 * hf) * CDIM + hoff + j * 16 + l16] = (bf16)val;
    }
}

// ---------------------------------------------------------------------------
// bias-gradient column sums + bias update: bout = (1-alpha)*bin - theta*colsum
// ---------------------------------------------------------------------------
__global__ void colsum_upd_kernel(const bf16* __restrict__ X,
                                  const float* __restrict__ bin,
                                  float* __restrict__ bout, int rows, int cols,
                                  const float* __restrict__ alpha_p,
                                  const float* __restrict__ theta_p) {
  int c = blockIdx.x * blockDim.x + threadIdx.x;
  if (c >= cols) return;
  float s = 0.f;
  for (int r = 0; r < rows; ++r) s += (float)X[(size_t)r * cols + c];
  bout[c] = (1.f - alpha_p[0]) * bin[c] - theta_p[0] * s;
}

// ---------------------------------------------------------------------------
// final sigmoid-gated merge; drops the persistent-memory rows
// ---------------------------------------------------------------------------
__global__ void merge_kernel(const float* __restrict__ z1,
                             const float* __restrict__ z2,
                             const float* __restrict__ swa,
                             const float* __restrict__ omem,
                             float* __restrict__ out) {
  int i = blockIdx.x * blockDim.x + threadIdx.x;
  if (i >= CB * CS * CDIM) return;
  int d = i % CDIM;
  int rem = i / CDIM;
  int s = rem % CS;
  int b = rem / CS;
  size_t src = ((size_t)(b * CT + CPM + s)) * CDIM + d;
  float g1 = 1.f / (1.f + __expf(-z1[src]));
  float g2 = 1.f / (1.f + __expf(-z2[src]));
  out[i] = g1 * swa[src] + g2 * omem[src];
}

// ---------------------------------------------------------------------------
// host-side orchestration
// ---------------------------------------------------------------------------
extern "C" void kernel_launch(void* const* d_in, const int* in_sizes, int n_in,
                              void* d_out, int out_size, void* d_ws,
                              size_t ws_size, hipStream_t stream) {
  (void)in_sizes; (void)n_in; (void)out_size; (void)ws_size;
  const float* x    = (const float*)d_in[0];
  const float* pmem = (const float*)d_in[1];
  const float* wq   = (const float*)d_in[2];
  const float* wk   = (const float*)d_in[3];
  const float* wv   = (const float*)d_in[4];
  const float* wo   = (const float*)d_in[5];
  const float* w1   = (const float*)d_in[6];
  const float* b1   = (const float*)d_in[7];
  const float* w2   = (const float*)d_in[8];
  const float* b2   = (const float*)d_in[9];
  const float* lin1 = (const float*)d_in[10];
  const float* lin2 = (const float*)d_in[11];
  const float* alpha = (const float*)d_in[12];
  const float* theta = (const float*)d_in[13];
  float* out = (float*)d_out;

  char* wsp = (char*)d_ws;
  auto alloc = [&](size_t bytes) -> char* {
    char* p = wsp;
    wsp += (bytes + 255) & ~(size_t)255;
    return p;
  };
  const size_t ND = (size_t)CN * CDIM;
  const size_t NH = (size_t)CN * CHID;
  const size_t DD = (size_t)CDIM * CDIM;
  const size_t HD = (size_t)CHID * CDIM;

  bf16* xin    = (bf16*)alloc(ND * 2);
  bf16* wq_b   = (bf16*)alloc(DD * 2);
  bf16* wk_b   = (bf16*)alloc(DD * 2);
  bf16* wv_b   = (bf16*)alloc(DD * 2);
  bf16* wo_b   = (bf16*)alloc(DD * 2);
  bf16* l1_b   = (bf16*)alloc(DD * 2);
  bf16* l2_b   = (bf16*)alloc(DD * 2);
  bf16* w1_b   = (bf16*)alloc(HD * 2);
  bf16* w2_b   = (bf16*)alloc(HD * 2);
  bf16* w1u_b  = (bf16*)alloc(HD * 2);
  bf16* w2u_b  = (bf16*)alloc(HD * 2);
  bf16* qbuf   = (bf16*)alloc(ND * 2);
  bf16* kbuf   = (bf16*)alloc(ND * 2);
  bf16* vbuf   = (bf16*)alloc(ND * 2);
  bf16* vtbuf  = (bf16*)alloc(ND * 2);
  bf16* attnb  = (bf16*)alloc(ND * 2);
  bf16* swa_b  = (bf16*)alloc(ND * 2);
  bf16* Eb     = (bf16*)alloc(ND * 2);
  bf16* omem_b = (bf16*)alloc(ND * 2);
  float* swa_f  = (float*)alloc(ND * 4);
  float* omem_f = (float*)alloc(ND * 4);
  float* z1 = (float*)alloc(ND * 4);
  float* z2 = (float*)alloc(ND * 4);
  bf16* H1  = (bf16*)alloc(NH * 2);
  bf16* A1  = (bf16*)alloc(NH * 2);
  bf16* DHb = (bf16*)alloc(NH * 2);
  float* b1u = (float*)alloc(CHID * 4);
  float* b2u = (float*)alloc(CDIM * 4);

  dim3 blk(256);
  const int dd = CDIM * CDIM, hd = CHID * CDIM;
  cvt_f32_bf16<<<(dd + 255) / 256, blk, 0, stream>>>(wq, wq_b, dd);
  cvt_f32_bf16<<<(dd + 255) / 256, blk, 0, stream>>>(wk, wk_b, dd);
  cvt_f32_bf16<<<(dd + 255) / 256, blk, 0, stream>>>(wv, wv_b, dd);
  cvt_f32_bf16<<<(dd + 255) / 256, blk, 0, stream>>>(wo, wo_b, dd);
  cvt_f32_bf16<<<(dd + 255) / 256, blk, 0, stream>>>(lin1, l1_b, dd);
  cvt_f32_bf16<<<(dd + 255) / 256, blk, 0, stream>>>(lin2, l2_b, dd);
  cvt_f32_bf16<<<(hd + 255) / 256, blk, 0, stream>>>(w1, w1_b, hd);
  cvt_f32_bf16<<<(hd + 255) / 256, blk, 0, stream>>>(w2, w2_b, hd);
  prep_xin_kernel<<<(CN * CDIM + 255) / 256, blk, 0, stream>>>(x, pmem, xin);

  dim3 gProj(CDIM / 256, CN / 64);   // (4, 66)
  dim3 gH(CHID / 256, CN / 64);      // (16, 66)
  dim3 gW1(CDIM / 256, CHID / 64);   // (4, 64)
  dim3 gW2(CHID / 256, CDIM / 64);   // (16, 16)

  // q/k/v = x_in @ W^T   -> bf16 [CN, CDIM]
  gemm_bf16_kernel<0, 1, 1><<<gProj, blk, 0, stream>>>(
      xin, wq_b, nullptr, qbuf, nullptr, nullptr, nullptr, nullptr, nullptr,
      nullptr, CN, CDIM, CDIM, CDIM, CDIM);
  gemm_bf16_kernel<0, 1, 1><<<gProj, blk, 0, stream>>>(
      xin, wk_b, nullptr, kbuf, nullptr, nullptr, nullptr, nullptr, nullptr,
      nullptr, CN, CDIM, CDIM, CDIM, CDIM);
  gemm_bf16_kernel<0, 1, 1><<<gProj, blk, 0, stream>>>(
      xin, wv_b, nullptr, vbuf, nullptr, nullptr, nullptr, nullptr, nullptr,
      nullptr, CN, CDIM, CDIM, CDIM, CDIM);
  vtrans_kernel<<<(CN * CDIM + 255) / 256, blk, 0, stream>>>(vbuf, vtbuf);
  // sliding-window attention
  flash_attn_kernel<<<dim3(CT / 64, CB * CNH), dim3(128), 0, stream>>>(
      qbuf, kbuf, vtbuf, attnb);
  // out_swa = attn @ wo^T   (f32 + bf16)
  gemm_bf16_kernel<0, 1, 2><<<gProj, blk, 0, stream>>>(
      attnb, wo_b, swa_f, swa_b, nullptr, nullptr, nullptr, nullptr, nullptr,
      nullptr, CN, CDIM, CDIM, CDIM, CDIM);
  // h = k @ w1^T + b1 ; a = silu(h)
  gemm_bf16_kernel<0, 1, 3><<<gH, blk, 0, stream>>>(
      xin, w1_b, nullptr, A1, H1, b1, nullptr, nullptr, nullptr, nullptr, CN,
      CHID, CDIM, CDIM, CDIM);
  // e = (a @ w2^T + b2) - out_swa
  gemm_bf16_kernel<0, 1, 4><<<gProj, blk, 0, stream>>>(
      A1, w2_b, nullptr, Eb, nullptr, b2, swa_f, nullptr, nullptr, nullptr, CN,
      CDIM, CHID, CHID, CHID);
  // dh = (e @ w2) * silu'(h)
  gemm_bf16_kernel<0, 0, 5><<<gH, blk, 0, stream>>>(
      Eb, w2_b, nullptr, DHb, nullptr, nullptr, nullptr, H1, nullptr, nullptr,
      CN, CHID, CDIM, CDIM, CHID);
  // w1' = (1-alpha)w1 - theta * dh^T @ k
  gemm_bf16_kernel<1, 0, 6><<<gW1, blk, 0, stream>>>(
      DHb, xin, nullptr, w1u_b, nullptr, nullptr, w1, nullptr, alpha, theta,
      CHID, CDIM, CN, CHID, CDIM);
  colsum_upd_kernel<<<CHID / 256, blk, 0, stream>>>(DHb, b1, b1u, CN, CHID,
                                                    alpha, theta);
  // w2' = (1-alpha)w2 - theta * e^T @ a
  gemm_bf16_kernel<1, 0, 6><<<gW2, blk, 0, stream>>>(
      Eb, A1, nullptr, w2u_b, nullptr, nullptr, w2, nullptr, alpha, theta,
      CDIM, CHID, CN, CDIM, CHID);
  colsum_upd_kernel<<<CDIM / 256, blk, 0, stream>>>(Eb, b2, b2u, CN, CDIM,
                                                    alpha, theta);
  // a2 = silu(k @ w1'^T + b1')
  gemm_bf16_kernel<0, 1, 7><<<gH, blk, 0, stream>>>(
      xin, w1u_b, nullptr, A1, nullptr, b1u, nullptr, nullptr, nullptr,
      nullptr, CN, CHID, CDIM, CDIM, CDIM);
  // out_mem = a2 @ w2'^T + b2'   (f32 + bf16)
  gemm_bf16_kernel<0, 1, 2><<<gProj, blk, 0, stream>>>(
      A1, w2u_b, omem_f, omem_b, nullptr, b2u, nullptr, nullptr, nullptr,
      nullptr, CN, CDIM, CHID, CHID, CHID);
  // gate logits
  gemm_bf16_kernel<0, 1, 0><<<gProj, blk, 0, stream>>>(
      swa_b, l1_b, z1, nullptr, nullptr, nullptr, nullptr, nullptr, nullptr,
      nullptr, CN, CDIM, CDIM, CDIM, CDIM);
  gemm_bf16_kernel<0, 1, 0><<<gProj, blk, 0, stream>>>(
      omem_b, l2_b, z2, nullptr, nullptr, nullptr, nullptr, nullptr, nullptr,
      nullptr, CN, CDIM, CDIM, CDIM, CDIM);
  // merged output, PM rows dropped
  merge_kernel<<<(CB * CS * CDIM + 255) / 256, blk, 0, stream>>>(z1, z2, swa_f,
                                                                 omem_f, out);
}
